// RNN_29068338659428
// MI455X (gfx1250) — matrix operand order repaired and checked by reading, not compiled, not run on hardware
//
#include <hip/hip_runtime.h>
#include <cstddef>

// ---------------------------------------------------------------------------
// Types for CDNA5 WMMA (wave32): bf16 16x16x32, f32 accumulate.
// ---------------------------------------------------------------------------
typedef __attribute__((ext_vector_type(16))) __bf16 bf16x16;
typedef __attribute__((ext_vector_type(8)))  __bf16 bf16x8;
typedef __attribute__((ext_vector_type(8)))  float  f32x8;

union FragU { bf16x16 v; bf16x8 h[2]; };

__device__ __forceinline__ __bf16 f2bf(float f) {
  unsigned u = __builtin_bit_cast(unsigned, f);
  unsigned r = u + 0x7FFFu + ((u >> 16) & 1u);   // round-to-nearest-even
  unsigned short hs = (unsigned short)(r >> 16);
  return __builtin_bit_cast(__bf16, hs);
}

// A-matrix fragment (16x32 bf16, row-major source, row stride = ldk elements).
// ISA layout: lanes 0-15 -> rows, K {0..7,16..23}; lanes 16-31 -> K {8..15,24..31}.
__device__ __forceinline__ bf16x16 load_a_frag(const __bf16* __restrict__ A,
                                               int ldk, int lane) {
  int m  = lane & 15;
  int kq = (lane >> 4) << 3;                    // 0 or 8
  const __bf16* p = A + (size_t)m * ldk + kq;
  FragU f;
  f.h[0] = *(const bf16x8*)(p);                 // K = kq .. kq+7
  f.h[1] = *(const bf16x8*)(p + 16);            // K = kq+16 .. kq+23
  return f.v;
}

// B-matrix fragment (32x16). B = W^T with W [N,K] row-major: K contiguous.
// ISA layout: lanes 0-15 -> n, K {0..7,16..23}; lanes 16-31 -> n, K {8..15,24..31}.
__device__ __forceinline__ bf16x16 load_b_frag(const __bf16* __restrict__ W,
                                               int ldk, int lane) {
  int n  = lane & 15;
  int kb = (lane >> 4) << 3;                    // 0 or 8
  const __bf16* p = W + (size_t)n * ldk + kb;
  FragU f;
  f.h[0] = *(const bf16x8*)(p);
  f.h[1] = *(const bf16x8*)(p + 16);
  return f.v;
}

__device__ __forceinline__ f32x8 wmma_bf16(bf16x16 a, bf16x16 b, f32x8 c) {
  return __builtin_amdgcn_wmma_f32_16x16x32_bf16(false, a, false, b, (short)0,
                                                 c, false, false);
}

// ---------------------------------------------------------------------------
// f32 -> bf16 conversion
// ---------------------------------------------------------------------------
__global__ void cvt_f32_to_bf16(const float* __restrict__ in,
                                __bf16* __restrict__ out, size_t n) {
  size_t i = (size_t)blockIdx.x * blockDim.x + threadIdx.x;
  size_t stride = (size_t)gridDim.x * blockDim.x;
  for (; i < n; i += stride) out[i] = f2bf(in[i]);
}

// ---------------------------------------------------------------------------
// C[M,N] = A[M,K](bf16) @ W[N,K]^T(bf16) + bias[N]   (f32 accumulate/output)
// Block: 256 threads = 8 waves; WG tile 256(M) x 64(N); wave = 32x64 strip.
// Ping-pong buffers (unroll-by-2), no loop-carried fragment copies.
// ---------------------------------------------------------------------------
__global__ __launch_bounds__(256)
void gemm_bf16_nt(const __bf16* __restrict__ A, const __bf16* __restrict__ W,
                  const float* __restrict__ bias, float* __restrict__ C,
                  int N, int K) {
  const int lane = threadIdx.x & 31;
  const int w    = threadIdx.x >> 5;
  const int m0   = blockIdx.x * 256 + w * 32;
  const int n0   = blockIdx.y * 64;

  f32x8 acc[2][4] = {};
  const __bf16* A0 = A + (size_t)m0 * K;
  const __bf16* A1 = A + (size_t)(m0 + 16) * K;
  const __bf16* Wp[4];
#pragma unroll
  for (int nt = 0; nt < 4; ++nt) Wp[nt] = W + (size_t)(n0 + nt * 16) * K;

  bf16x16 pa0, pa1, pb[4];   // ping
  bf16x16 qa0, qa1, qb[4];   // pong

  // ping <- k = 0
  pa0 = load_a_frag(A0, K, lane);
  pa1 = load_a_frag(A1, K, lane);
#pragma unroll
  for (int nt = 0; nt < 4; ++nt) pb[nt] = load_b_frag(Wp[nt], K, lane);

  int k0 = 0;
  for (; k0 + 64 < K; k0 += 64) {
    // pong <- k0+32
    qa0 = load_a_frag(A0 + k0 + 32, K, lane);
    qa1 = load_a_frag(A1 + k0 + 32, K, lane);
#pragma unroll
    for (int nt = 0; nt < 4; ++nt) qb[nt] = load_b_frag(Wp[nt] + k0 + 32, K, lane);
    // compute ping (k0)
#pragma unroll
    for (int nt = 0; nt < 4; ++nt) {
      acc[0][nt] = wmma_bf16(pa0, pb[nt], acc[0][nt]);
      acc[1][nt] = wmma_bf16(pa1, pb[nt], acc[1][nt]);
    }
    // ping <- k0+64
    pa0 = load_a_frag(A0 + k0 + 64, K, lane);
    pa1 = load_a_frag(A1 + k0 + 64, K, lane);
#pragma unroll
    for (int nt = 0; nt < 4; ++nt) pb[nt] = load_b_frag(Wp[nt] + k0 + 64, K, lane);
    // compute pong (k0+32)
#pragma unroll
    for (int nt = 0; nt < 4; ++nt) {
      acc[0][nt] = wmma_bf16(qa0, qb[nt], acc[0][nt]);
      acc[1][nt] = wmma_bf16(qa1, qb[nt], acc[1][nt]);
    }
  }
  // tail: ping holds K-64; pong <- K-32
  qa0 = load_a_frag(A0 + k0 + 32, K, lane);
  qa1 = load_a_frag(A1 + k0 + 32, K, lane);
#pragma unroll
  for (int nt = 0; nt < 4; ++nt) qb[nt] = load_b_frag(Wp[nt] + k0 + 32, K, lane);
#pragma unroll
  for (int nt = 0; nt < 4; ++nt) {
    acc[0][nt] = wmma_bf16(pa0, pb[nt], acc[0][nt]);
    acc[1][nt] = wmma_bf16(pa1, pb[nt], acc[1][nt]);
  }
#pragma unroll
  for (int nt = 0; nt < 4; ++nt) {
    acc[0][nt] = wmma_bf16(qa0, qb[nt], acc[0][nt]);
    acc[1][nt] = wmma_bf16(qa1, qb[nt], acc[1][nt]);
  }

  const int row_off = (lane >> 4) << 3;
  const int col_l   = lane & 15;
#pragma unroll
  for (int mt = 0; mt < 2; ++mt) {
#pragma unroll
    for (int nt = 0; nt < 4; ++nt) {
      int col = n0 + nt * 16 + col_l;
      float bv = bias[col];
#pragma unroll
      for (int r = 0; r < 8; ++r) {
        int row = m0 + mt * 16 + row_off + r;
        C[(size_t)row * N + col] = acc[mt][nt][r] + bv;
      }
    }
  }
}

// ---------------------------------------------------------------------------
// Grid-wide barrier (all 48 WGs resident on MI455X)
// ---------------------------------------------------------------------------
__device__ __forceinline__ void grid_sync(unsigned* cnt, unsigned bar,
                                          unsigned nwg) {
  __builtin_amdgcn_fence(__ATOMIC_RELEASE, "agent");
  __syncthreads();
  if (threadIdx.x == 0) {
    __hip_atomic_fetch_add(cnt, 1u, __ATOMIC_RELAXED, __HIP_MEMORY_SCOPE_AGENT);
    while (__hip_atomic_load(cnt, __ATOMIC_RELAXED, __HIP_MEMORY_SCOPE_AGENT) <
           bar * nwg) {
      __builtin_amdgcn_s_sleep(1);
    }
  }
  __syncthreads();
  __builtin_amdgcn_fence(__ATOMIC_ACQUIRE, "agent");
}

__global__ void init_state(float* __restrict__ h, __bf16* __restrict__ hbf,
                           unsigned* __restrict__ cnt, int n) {
  int i = blockIdx.x * blockDim.x + threadIdx.x;
  if (i < n) { h[i] = 0.0f; hbf[i] = f2bf(0.0f); }
  if (i == 0) *cnt = 0u;
}

// ---------------------------------------------------------------------------
// Persistent GRU recurrence for one layer. 48 WGs x 256 threads.
// Phase 1 (all WGs): hg[64,3072] tile = hbf[64,1024] @ Whh^T + bhh  (WMMA)
// Phase 2 (all WGs): gates + h update + residual-add output
// ---------------------------------------------------------------------------
__global__ __launch_bounds__(256)
void gru_recurrent(const float* __restrict__ xg,   // [T,B,3H]
                   const float* __restrict__ res,  // [T,B,H]
                   const __bf16* __restrict__ Whh, // [3H,H] bf16
                   const float* __restrict__ bhh,  // [3H]
                   float* __restrict__ h,          // [B,H]
                   __bf16* __restrict__ hbf,       // [B,H]
                   float* __restrict__ hg,         // [B,3H]
                   __bf16* outbf,                  // [T,B,H] or null
                   float* outf32,                  // [T,B,H] or null
                   float* __restrict__ hidden,     // [B,H]
                   unsigned* __restrict__ counter, int T) {
  const int Bb = 64, H = 1024, G = 3072;
  const int lane = threadIdx.x & 31;
  const int w    = threadIdx.x >> 5;
  const int mi   = w >> 1;            // 0..3  (16-row subtile of B=64)
  const int nj0  = (w & 1) * 2;       // 0 or 2 (two 16-col subtiles)
  const int n0   = blockIdx.x * 64;   // column tile within 3H
  const unsigned nwg = gridDim.x;
  unsigned bar = 0;

  const int row_off = (lane >> 4) << 3;
  const int col_l   = lane & 15;
  const int lid     = (int)blockIdx.x * 256 + (int)threadIdx.x;  // 0..12287
  const int nlid    = (int)nwg * 256;                            // 12288
  const __bf16* W0 = Whh + (size_t)(n0 + nj0 * 16) * H;
  const __bf16* W1 = Whh + (size_t)(n0 + (nj0 + 1) * 16) * H;

  for (int t = 0; t < T; ++t) {
    // Warm L2 with this step's xg/res slices while the WMMA loop runs.
    {
      const char* px = (const char*)(xg + (size_t)t * Bb * G);
      __builtin_prefetch(px + (size_t)lid * 64, 0, 1);   // 12288*64B = 786KB
      if (lid < (Bb * H * 4) / 64) {
        const char* pr = (const char*)(res + (size_t)t * Bb * H);
        __builtin_prefetch(pr + (size_t)lid * 64, 0, 1); // 4096*64B = 256KB
      }
    }

    // ---- Phase 1: recurrent GEMM tile (ping-pong pipelined) ----
    f32x8 acc[2] = {};
    const __bf16* Arow = hbf + (size_t)(mi * 16) * H;

    bf16x16 pa, pb0, pb1, qa, qb0, qb1;
    pa  = load_a_frag(Arow, H, lane);
    pb0 = load_b_frag(W0, H, lane);
    pb1 = load_b_frag(W1, H, lane);
    int k0 = 0;
    for (; k0 + 64 < H; k0 += 64) {
      qa  = load_a_frag(Arow + k0 + 32, H, lane);
      qb0 = load_b_frag(W0 + k0 + 32, H, lane);
      qb1 = load_b_frag(W1 + k0 + 32, H, lane);
      acc[0] = wmma_bf16(pa, pb0, acc[0]);
      acc[1] = wmma_bf16(pa, pb1, acc[1]);
      pa  = load_a_frag(Arow + k0 + 64, H, lane);
      pb0 = load_b_frag(W0 + k0 + 64, H, lane);
      pb1 = load_b_frag(W1 + k0 + 64, H, lane);
      acc[0] = wmma_bf16(qa, qb0, acc[0]);
      acc[1] = wmma_bf16(qa, qb1, acc[1]);
    }
    qa  = load_a_frag(Arow + k0 + 32, H, lane);
    qb0 = load_b_frag(W0 + k0 + 32, H, lane);
    qb1 = load_b_frag(W1 + k0 + 32, H, lane);
    acc[0] = wmma_bf16(pa, pb0, acc[0]);
    acc[1] = wmma_bf16(pa, pb1, acc[1]);
    acc[0] = wmma_bf16(qa, qb0, acc[0]);
    acc[1] = wmma_bf16(qa, qb1, acc[1]);

#pragma unroll
    for (int j = 0; j < 2; ++j) {
      int col = n0 + (nj0 + j) * 16 + col_l;
      float bv = bhh[col];
#pragma unroll
      for (int r = 0; r < 8; ++r) {
        int row = mi * 16 + row_off + r;
        hg[(size_t)row * G + col] = acc[j][r] + bv;
      }
    }
    grid_sync(counter, ++bar, nwg);

    // ---- Phase 2: gates, spread over all 48 WGs ----
    {
      const float* xgt  = xg  + (size_t)t * Bb * G;
      const float* rest = res + (size_t)t * Bb * H;
      const int total = Bb * H;                 // 65536
      for (int i = lid; i < total; i += nlid) {
        int b = i >> 10;
        int j = i & 1023;
        float xr = xgt[(size_t)b * G + j];
        float xz = xgt[(size_t)b * G + H + j];
        float xn = xgt[(size_t)b * G + 2 * H + j];
        float hr = hg[(size_t)b * G + j];
        float hz = hg[(size_t)b * G + H + j];
        float hn = hg[(size_t)b * G + 2 * H + j];
        float rg = 1.0f / (1.0f + expf(-(xr + hr)));
        float zg = 1.0f / (1.0f + expf(-(xz + hz)));
        float ng = tanhf(xn + rg * hn);
        float hprev = h[i];
        float hnew = (1.0f - zg) * ng + zg * hprev;
        h[i]   = hnew;
        hbf[i] = f2bf(hnew);
        float o = hnew + rest[i];
        if (outbf)  outbf[(size_t)t * Bb * H + i] = f2bf(o);
        if (outf32) outf32[(size_t)t * Bb * H + i] = o;
        if (t == T - 1) hidden[i] = hnew;
      }
    }
    grid_sync(counter, ++bar, nwg);
  }
}

// ---------------------------------------------------------------------------
// Host-side orchestration
// ---------------------------------------------------------------------------
extern "C" void kernel_launch(void* const* d_in, const int* in_sizes, int n_in,
                              void* d_out, int out_size, void* d_ws,
                              size_t ws_size, hipStream_t stream) {
  (void)in_sizes; (void)n_in; (void)out_size; (void)ws_size;
  const float* X    = (const float*)d_in[0];
  const float* Wih0 = (const float*)d_in[1];
  const float* Whh0 = (const float*)d_in[2];
  const float* bih0 = (const float*)d_in[3];
  const float* bhh0 = (const float*)d_in[4];
  const float* Wd0  = (const float*)d_in[5];
  const float* bd0  = (const float*)d_in[6];
  const float* Wih1 = (const float*)d_in[7];
  const float* Whh1 = (const float*)d_in[8];
  const float* bih1 = (const float*)d_in[9];
  const float* bhh1 = (const float*)d_in[10];
  const float* Wd1  = (const float*)d_in[11];
  const float* bd1  = (const float*)d_in[12];

  const int T = 512, Bb = 64, D = 1024, H = 1024, G = 3 * H;
  const size_t M = (size_t)T * Bb;            // 32768 rows

  // workspace carve-out (256B aligned)
  char* base = (char*)d_ws;
  size_t off = 0;
  auto alloc = [&](size_t bytes) {
    void* p = base + off;
    off = (off + bytes + 255) & ~(size_t)255;
    return p;
  };
  __bf16* Xbf   = (__bf16*)alloc(M * D * 2);        // X in bf16
  __bf16* X1bf  = (__bf16*)alloc(M * H * 2);        // layer-1 input, bf16
  __bf16* WihB  = (__bf16*)alloc((size_t)G * D * 2);
  __bf16* WhhB  = (__bf16*)alloc((size_t)G * H * 2);
  __bf16* WdB   = (__bf16*)alloc((size_t)H * D * 2);
  float*  xgbuf = (float*)alloc(M * G * 4);         // input-side gates
  float*  resb  = (float*)alloc(M * H * 4);         // dense residual
  float*  hgbuf = (float*)alloc((size_t)Bb * G * 4);
  float*  hbuf  = (float*)alloc((size_t)Bb * H * 4);
  __bf16* hbf   = (__bf16*)alloc((size_t)Bb * H * 2);
  unsigned* cnt = (unsigned*)alloc(256);

  float* out    = (float*)d_out;
  float* hid0   = out + M * H;          // hidden_state layer 0
  float* hid1   = hid0 + (size_t)Bb * H;

  dim3 blk(256);
  auto cvt = [&](const float* src, __bf16* dst, size_t n) {
    int g = (int)((n + 255) / 256);
    if (g > 4096) g = 4096;
    cvt_f32_to_bf16<<<g, blk, 0, stream>>>(src, dst, n);
  };

  // ---------------- Layer 0 ----------------
  cvt(X, Xbf, M * D);
  cvt(Wih0, WihB, (size_t)G * D);
  cvt(Whh0, WhhB, (size_t)G * H);
  cvt(Wd0, WdB, (size_t)H * D);

  gemm_bf16_nt<<<dim3((unsigned)(M / 256), G / 64), blk, 0, stream>>>(
      Xbf, WihB, bih0, xgbuf, G, D);
  gemm_bf16_nt<<<dim3((unsigned)(M / 256), H / 64), blk, 0, stream>>>(
      Xbf, WdB, bd0, resb, H, D);

  init_state<<<(Bb * H + 255) / 256, blk, 0, stream>>>(hbuf, hbf, cnt, Bb * H);
  gru_recurrent<<<G / 64, blk, 0, stream>>>(xgbuf, resb, WhhB, bhh0, hbuf, hbf,
                                            hgbuf, X1bf, nullptr, hid0, cnt, T);

  // ---------------- Layer 1 ----------------
  cvt(Wih1, WihB, (size_t)G * H);
  cvt(Whh1, WhhB, (size_t)G * H);
  cvt(Wd1, WdB, (size_t)H * H);

  gemm_bf16_nt<<<dim3((unsigned)(M / 256), G / 64), blk, 0, stream>>>(
      X1bf, WihB, bih1, xgbuf, G, H);
  gemm_bf16_nt<<<dim3((unsigned)(M / 256), H / 64), blk, 0, stream>>>(
      X1bf, WdB, bd1, resb, H, H);

  init_state<<<(Bb * H + 255) / 256, blk, 0, stream>>>(hbuf, hbf, cnt, Bb * H);
  gru_recurrent<<<G / 64, blk, 0, stream>>>(xgbuf, resb, WhhB, bhh1, hbuf, hbf,
                                            hgbuf, nullptr, out, hid1, cnt, T);
}
